// AlwGAT_49898930045026
// MI455X (gfx1250) — compile-verified, matrix operand order checked by reading, not run on hardware
//
#include <hip/hip_runtime.h>
#include <stdint.h>

// Problem constants (from reference)
#define SELF_DIM 36
#define OTHER_DIM 28
#define OTHER_NUM 19
#define HID 64
#define OBS 568   // 36 + 28*19

typedef _Float16 v16h __attribute__((ext_vector_type(16)));
typedef float    v8f  __attribute__((ext_vector_type(8)));

union Frag { v16h v; uint4 q[2]; };

// ---------------------------------------------------------------------------
// Setup kernel: build fused weights.
//   out[n] = self(36) @ Wself + oatt(28) @ Woth + bfused[n]
//   Wself[i][n] = sum_h We[i][h]*Wo[h][n] + (Ws[i][h]+Ws[i+36][h])*Wo[64+h][n]
//   Woth [k][n] = sum_h We[36+k][h]*Wo[h][n]
// Stored transposed (n-major, fp16) so WMMA B-fragment loads are contiguous.
// ---------------------------------------------------------------------------
__global__ __launch_bounds__(256) void fuse_weights(
    const float* __restrict__ We, const float* __restrict__ be,
    const float* __restrict__ Ws, const float* __restrict__ bs,
    const float* __restrict__ Wo, const float* __restrict__ bo,
    _Float16* __restrict__ WfT, float* __restrict__ bfused)
{
  const int tid = threadIdx.x;
  for (int idx = tid; idx < 64 * 64; idx += 256) {
    const int k = idx >> 6;     // fused input feature (0..63)
    const int n = idx & 63;     // output channel
    float acc = 0.f;
    if (k < SELF_DIM) {
      for (int h = 0; h < HID; ++h)
        acc += We[k * HID + h] * Wo[h * HID + n]
             + (Ws[k * HID + h] + Ws[(k + SELF_DIM) * HID + h]) * Wo[(HID + h) * HID + n];
    } else {
      const int kk = k - SELF_DIM;  // 0..27
      for (int h = 0; h < HID; ++h)
        acc += We[(SELF_DIM + kk) * HID + h] * Wo[h * HID + n];
    }
    WfT[n * 64 + k] = (_Float16)acc;
  }
  if (tid < 64) {
    float b = bo[tid];
    for (int h = 0; h < HID; ++h)
      b += be[h] * Wo[h * HID + tid] + bs[h] * Wo[(HID + h) * HID + tid];
    bfused[tid] = b;
  }
}

// ---------------------------------------------------------------------------
// Main kernel: 256 threads/block, 1 row per thread (wave32 -> 32 rows/wave).
// Phase 1 (VALU): stream the row once; online-softmax attention pooling into
//                 a 64-dim feature vector [self(36) ; oatt(28)], staged to LDS
//                 as fp16.
// Phase 2 (WMMA): each wave multiplies its two 16x64 feature tiles against the
//                 fused 64x64 weight with v_wmma_f32_16x16x32_f16 (2 k-steps x
//                 4 n-tiles), adds fused bias, stores fp32 output.
// ---------------------------------------------------------------------------
__global__ __launch_bounds__(256) void obs_encoder(
    const float* __restrict__ x,
    const float* __restrict__ Wa, const float* __restrict__ ba,
    const _Float16* __restrict__ WfT, const float* __restrict__ bfused,
    float* __restrict__ out)
{
  __shared__ _Float16 feat[256 * 64];   // 32 KB: one 64-dim fp16 vector per row
  const int tid = threadIdx.x;
  const int row = blockIdx.x * 256 + tid;
  const float* xr = x + (size_t)row * OBS;

  // ---- self part + logit base: sdot = self . Wa[:36] + ba ----
  float selfv[SELF_DIM];
  const float4* xv = (const float4*)xr;            // rows are 16B-aligned (568*4 % 16 == 0)
  #pragma unroll
  for (int i = 0; i < SELF_DIM / 4; ++i) {
    float4 q = xv[i];
    selfv[4 * i + 0] = q.x; selfv[4 * i + 1] = q.y;
    selfv[4 * i + 2] = q.z; selfv[4 * i + 3] = q.w;
  }
  float sdot = ba[0];
  #pragma unroll
  for (int i = 0; i < SELF_DIM; ++i) sdot += selfv[i] * Wa[i];

  unsigned* featp = (unsigned*)&feat[tid * 64];    // packed fp16 pair writes
  #pragma unroll
  for (int i = 0; i < SELF_DIM / 2; ++i) {
    union { _Float16 h[2]; unsigned u; } p;
    p.h[0] = (_Float16)selfv[2 * i];
    p.h[1] = (_Float16)selfv[2 * i + 1];
    featp[i] = p.u;
  }

  // ---- online-softmax weighted pooling over the 19 others (single pass) ----
  float m = -3.0e38f, ssum = 0.f;
  float vacc[OTHER_DIM];
  #pragma unroll
  for (int k = 0; k < OTHER_DIM; ++k) vacc[k] = 0.f;

  for (int j = 0; j < OTHER_NUM; ++j) {
    float oth[OTHER_DIM];
    const float4* ov = (const float4*)(xr + SELF_DIM + j * OTHER_DIM);  // 16B aligned
    #pragma unroll
    for (int i = 0; i < OTHER_DIM / 4; ++i) {
      float4 q = ov[i];
      oth[4 * i + 0] = q.x; oth[4 * i + 1] = q.y;
      oth[4 * i + 2] = q.z; oth[4 * i + 3] = q.w;
    }
    float l = sdot;
    #pragma unroll
    for (int k = 0; k < OTHER_DIM; ++k) l += oth[k] * Wa[SELF_DIM + k];
    // branchless online softmax update
    float mn   = fmaxf(m, l);
    float cold = __expf(m - mn);   // exp(-inf)=0 on first iteration
    float w    = __expf(l - mn);
    ssum = ssum * cold + w;
    #pragma unroll
    for (int k = 0; k < OTHER_DIM; ++k) vacc[k] = vacc[k] * cold + w * oth[k];
    m = mn;
  }
  const float inv = 1.0f / ssum;
  #pragma unroll
  for (int k = 0; k < OTHER_DIM / 2; ++k) {
    union { _Float16 h[2]; unsigned u; } p;
    p.h[0] = (_Float16)(vacc[2 * k]     * inv);
    p.h[1] = (_Float16)(vacc[2 * k + 1] * inv);
    featp[SELF_DIM / 2 + k] = p.u;
  }
  __syncthreads();

  // ---- WMMA phase ----
  const int wave = tid >> 5, lane = tid & 31;
  const int ln = lane & 15, half = lane >> 4;

  // B fragments: [n-tile 0..3][k-step 0..1]; fp16 A-layout-symmetric:
  //   first 8 halves: K = s*32 + half*8 + (0..7); next 8: K += 16
  Frag bf[4][2];
  #pragma unroll
  for (int tn = 0; tn < 4; ++tn) {
    const int n = tn * 16 + ln;
    #pragma unroll
    for (int s = 0; s < 2; ++s) {
      bf[tn][s].q[0] = *(const uint4*)(WfT + n * 64 + s * 32 + half * 8);
      bf[tn][s].q[1] = *(const uint4*)(WfT + n * 64 + s * 32 + 16 + half * 8);
    }
  }
  float bias[4];
  #pragma unroll
  for (int tn = 0; tn < 4; ++tn) bias[tn] = bfused[tn * 16 + ln];

  #pragma unroll
  for (int rt = 0; rt < 2; ++rt) {
    const int r = wave * 32 + rt * 16 + ln;   // LDS feature row for this lane's M
    Frag a[2];
    #pragma unroll
    for (int s = 0; s < 2; ++s) {
      a[s].q[0] = *(const uint4*)(&feat[r * 64 + s * 32 + half * 8]);
      a[s].q[1] = *(const uint4*)(&feat[r * 64 + s * 32 + 16 + half * 8]);
    }
    const int growBase = blockIdx.x * 256 + wave * 32 + rt * 16 + half * 8;
    #pragma unroll
    for (int tn = 0; tn < 4; ++tn) {
      v8f c = {};
      c = __builtin_amdgcn_wmma_f32_16x16x32_f16(false, a[0].v, false, bf[tn][0].v,
                                                 (short)0, c, false, false);
      c = __builtin_amdgcn_wmma_f32_16x16x32_f16(false, a[1].v, false, bf[tn][1].v,
                                                 (short)0, c, false, false);
      const int n = tn * 16 + ln;
      // C/D layout: VGPR v -> row (half*8 + v), col = n
      #pragma unroll
      for (int vv = 0; vv < 8; ++vv)
        out[(size_t)(growBase + vv) * HID + n] = c[vv] + bias[tn];
    }
  }
}

// ---------------------------------------------------------------------------
extern "C" void kernel_launch(void* const* d_in, const int* in_sizes, int n_in,
                              void* d_out, int out_size, void* d_ws, size_t ws_size,
                              hipStream_t stream) {
  const float* x  = (const float*)d_in[0];
  const float* Wa = (const float*)d_in[1];
  const float* ba = (const float*)d_in[2];
  const float* We = (const float*)d_in[3];
  const float* be = (const float*)d_in[4];
  const float* Ws = (const float*)d_in[5];
  const float* bs = (const float*)d_in[6];
  const float* Wo = (const float*)d_in[7];
  const float* bo = (const float*)d_in[8];
  float* out = (float*)d_out;

  _Float16* WfT  = (_Float16*)d_ws;                       // 64*64 fp16 = 8 KB
  float*    bfus = (float*)((char*)d_ws + 64 * 64 * 2);   // 64 fp32

  const int B = in_sizes[0] / OBS;                        // 65536
  fuse_weights<<<1, 256, 0, stream>>>(We, be, Ws, bs, Wo, bo, WfT, bfus);
  obs_encoder<<<B / 256, 256, 0, stream>>>(x, Wa, ba, WfT, bfus, out);
}
